// TopEncoderBase_52621939310836
// MI455X (gfx1250) — matrix operand order repaired and checked by reading, not compiled, not run on hardware
//
#include <hip/hip_runtime.h>
#include <hip/hip_bf16.h>

// ---------------------------------------------------------------------------
// MI455X (gfx1250) fused QKV-attention-ish reduction.
// Memory-bound: x is 268 MB f32 -> read exactly once (~11.5us floor @23.3TB/s).
// GEMM (65536x1024 @ 1024x2048) on v_wmma_f32_16x16x32_bf16, f32 accumulate.
// Wkv is pre-converted+transposed to bf16 (4 MB, L2-resident), so WMMA B
// fragments are per-lane contiguous global_load_b128 pairs -> no LDS staging
// of B, no per-K-step barriers. Everything after the GEMM (rope, q.k dots,
// v-norms, k-reduction) is fused in-LDS; kv intermediate never hits HBM.
// q row is staged global->LDS via the CDNA5 async path
// (global_load_async_to_lds_b32 + s_wait_asynccnt).
// ---------------------------------------------------------------------------

typedef __bf16 bf16;
typedef __attribute__((ext_vector_type(16))) __bf16 v16bf;
typedef __attribute__((ext_vector_type(8)))  __bf16 v8bf;
typedef __attribute__((ext_vector_type(4)))  __bf16 v4bf;
typedef __attribute__((ext_vector_type(8)))  float  v8f;

#define DH   16
#define DE   64
#define DIN  1024
#define NKV  2048   // 2*DH*DE output columns of Wkv
#define DK   2048
#define DZ   32
#define TM   32     // (z,k) rows per workgroup
#define KC   32     // K chunk per WMMA step

// --- LDS layout (bytes). A tile is unioned inside the kv staging region
// --- (kv staging written only after all WMMA reads of A complete).
#define A_STRIDE   (DIN + 8)   // bf16 units, padded vs bank conflicts
#define KV_STRIDE  (NKV + 4)   // f32 units
#define OFF_A      0
#define OFF_KV     0
#define SZ_KV      (TM * KV_STRIDE * 4)           // 262656
#define OFF_Q      262656
#define OFF_NUM    (OFF_Q + 4096)                 // 266752
#define OFF_DEN    (OFF_NUM + 4096)               // 270848
#define SMEM_TOTAL 270912

__global__ void zero_kernel(float* __restrict__ p, int n) {
    int i = blockIdx.x * 256 + threadIdx.x;
    if (i < n) p[i] = 0.0f;
}

// Wkv (1024x2048 f32, row-major) -> WkvT (2048x1024 bf16, i.e. transposed),
// via 32x32 LDS tiles so both global streams stay coalesced. One-time 8 MB.
__global__ __launch_bounds__(256)
void cvtT_kernel(const float* __restrict__ w, bf16* __restrict__ wt) {
    __shared__ bf16 tile[32][34];
    const int bx = blockIdx.x;            // col tile (2048/32 = 64)
    const int by = blockIdx.y;            // row tile (1024/32 = 32)
    const int tx = threadIdx.x & 31;
    const int ty = threadIdx.x >> 5;      // 0..7
    #pragma unroll
    for (int i = 0; i < 4; ++i) {
        int r = by * 32 + ty + i * 8;
        tile[ty + i * 8][tx] = (bf16)w[(size_t)r * NKV + bx * 32 + tx];
    }
    __syncthreads();
    #pragma unroll
    for (int i = 0; i < 4; ++i) {
        int c = bx * 32 + ty + i * 8;
        wt[(size_t)c * DIN + by * 32 + tx] = tile[tx][ty + i * 8];
    }
}

// q = x[:,0] @ Wq + bq   (32 x 1024, trivial)
__global__ __launch_bounds__(256)
void q_kernel(const float* __restrict__ x, const float* __restrict__ Wq,
              const float* __restrict__ bq, float* __restrict__ q_g) {
    __shared__ float xr[DIN];
    const int z   = blockIdx.x;
    const int col = blockIdx.y * 256 + threadIdx.x;
    for (int i = threadIdx.x; i < DIN; i += 256)
        xr[i] = x[(size_t)z * DK * DIN + i];
    __syncthreads();
    float acc = bq[col];
    for (int i = 0; i < DIN; ++i)
        acc = fmaf(xr[i], Wq[i * (DH * DE) + col], acc);
    q_g[z * (DH * DE) + col] = acc;
}

__global__ __launch_bounds__(1024)
void kv_attn_kernel(const float* __restrict__ x,
                    const unsigned char* __restrict__ mask,
                    const bf16* __restrict__ wkvT,   // [NKV][DIN] bf16
                    const float* __restrict__ bkv,
                    const float* __restrict__ q_g,
                    float* __restrict__ num_g,
                    float* __restrict__ den_g) {
    extern __shared__ char smem[];
    bf16*  As       = (bf16*)(smem + OFF_A);
    float* kvs      = (float*)(smem + OFF_KV);
    float* qs       = (float*)(smem + OFF_Q);
    float* num_part = (float*)(smem + OFF_NUM);
    float* den_part = (float*)(smem + OFF_DEN);

    const int tid     = threadIdx.x;
    const int lane    = tid & 31;
    const int wave    = tid >> 5;
    const int rowBase = blockIdx.x * TM;      // TM divides DK -> one z per WG
    const int z       = rowBase / DK;

    // ---- async-stage the q row (4 KB) global->LDS on the CDNA5 async path
    {
        float*       qdst = qs + tid;
        const float* qsrc = q_g + z * (DH * DE) + tid;
        unsigned     ldsb = (unsigned)(uintptr_t)qdst;  // low 32b of flat = LDS addr
        asm volatile("global_load_async_to_lds_b32 %0, %1, off"
                     :: "v"(ldsb), "v"(qsrc)
                     : "memory");
    }
    num_part[tid] = 0.0f;
    if (tid < DH) den_part[tid] = 0.0f;

    // ---- stage A tile: 32 rows of x, f32 -> bf16 in LDS (read x ONCE) ----
    {
        const float4* xv = (const float4*)(x + (size_t)rowBase * DIN);
        #pragma unroll
        for (int i = 0; i < 8; ++i) {
            int v = tid + i * 1024;            // 8192 float4 total
            float4 f = xv[v];
            int e = v * 4;
            int r = e >> 10;
            int c = e & (DIN - 1);
            v4bf p;
            p[0] = (bf16)f.x; p[1] = (bf16)f.y; p[2] = (bf16)f.z; p[3] = (bf16)f.w;
            *(v4bf*)(As + r * A_STRIDE + c) = p;
        }
    }
    asm volatile("s_wait_asynccnt 0" ::: "memory");
    __syncthreads();

    // ---- WMMA GEMM: each wave owns 16 rows x 128 cols (8 tiles) ----
    const int mg      = wave >> 4;             // 0..1  (M group of 16 rows)
    const int ng      = wave & 15;             // 0..15 (N group of 128 cols)
    const int colBase = ng * 128;
    const int mrow    = mg * 16 + (lane & 15);
    const int kbaseA  = (lane < 16) ? 0 : 8;   // ISA 16-bit A operand layout
    const int kbaseB  = (lane < 16) ? 0 : 8;   // ISA 16-bit B operand layout

    v8f acc[8];
    #pragma unroll
    for (int t = 0; t < 8; ++t) { v8f zz = {}; acc[t] = zz; }

    // per-lane base pointers for the 8 B-columns this lane touches
    const bf16* bcol = wkvT + (size_t)(colBase + (lane & 15)) * DIN + kbaseB;

    for (int ks = 0; ks < DIN / KC; ++ks) {
        // A fragment: lane holds row M=mrow, K = kbaseA+{0..7,16..23}
        const bf16* arow = As + mrow * A_STRIDE + ks * KC + kbaseA;
        v8bf a0 = *(const v8bf*)(arow);
        v8bf a1 = *(const v8bf*)(arow + 16);
        v16bf a = __builtin_shufflevector(a0, a1, 0, 1, 2, 3, 4, 5, 6, 7,
                                          8, 9, 10, 11, 12, 13, 14, 15);
        #pragma unroll
        for (int t = 0; t < 8; ++t) {
            // B fragment: K = kbaseB+{0..7,16..23} of this lane's column (L2-resident)
            const bf16* bp = bcol + (size_t)t * 16 * DIN + ks * KC;
            v8bf b0 = *(const v8bf*)(bp);
            v8bf b1 = *(const v8bf*)(bp + 16);
            v16bf b = __builtin_shufflevector(b0, b1, 0, 1, 2, 3, 4, 5, 6, 7,
                                              8, 9, 10, 11, 12, 13, 14, 15);
            acc[t] = __builtin_amdgcn_wmma_f32_16x16x32_bf16(
                false, a, false, b, (short)0, acc[t], false, false);
        }
    }

    // ---- bias add + stage kv tile in LDS (union with A region) ----
    __syncthreads();   // all WMMA reads of As done across the workgroup
    #pragma unroll
    for (int t = 0; t < 8; ++t) {
        int col    = colBase + t * 16 + (lane & 15);
        float bias = bkv[col];
        int rbase  = mg * 16 + ((lane < 16) ? 0 : 8);  // C layout: VGPR j -> M=j / j+8
        #pragma unroll
        for (int j = 0; j < 8; ++j)
            kvs[(rbase + j) * KV_STRIDE + col] = acc[t][j] + bias;
    }
    __syncthreads();

    // ---- fused post-processing: wave w owns row w of the tile ----
    {
        const int  gk     = rowBase + wave;
        const int  kpos   = gk & (DK - 1);
        const bool masked = mask[gk] != 0;
        const float* kv   = kvs + wave * KV_STRIDE;
        // rope: lane j handles rotation dim j (inv_freq = 10000^(-j/32))
        const float ang = (float)kpos *
                          __expf((float)lane * (-9.210340371976184f / 32.0f));
        const float cv = __cosf(ang);
        const float sv = __sinf(ang);

        #pragma unroll 4
        for (int h = 0; h < DH; ++h) {
            float q1 = qs[h * DE + lane];
            float q2 = qs[h * DE + 32 + lane];
            float k1 = kv[h * DE + lane];
            float k2 = kv[h * DE + 32 + lane];
            // q . rope(k) with rotation folded onto q
            float p = k1 * (q1 * cv + q2 * sv) + k2 * (q2 * cv - q1 * sv);
            #pragma unroll
            for (int off = 16; off >= 1; off >>= 1)
                p += __shfl_xor(p, off, 32);
            float s2 = masked ? 0.0f : p * p;

            float v1  = kv[DH * DE + h * DE + lane];
            float v2  = kv[DH * DE + h * DE + 32 + lane];
            float sq1 = v1 * v1, sq2 = v2 * v2;
            float vs  = sq1 + sq2;
            #pragma unroll
            for (int off = 16; off >= 1; off >>= 1)
                vs += __shfl_xor(vs, off, 32);
            float scale = s2 / fmaxf(vs, 1e-38f);
            atomicAdd(&num_part[h * DE + lane],      sq1 * scale);
            atomicAdd(&num_part[h * DE + 32 + lane], sq2 * scale);
            if (lane == 0) atomicAdd(&den_part[h], s2);
        }
    }
    __syncthreads();
    // one global flush per workgroup
    atomicAdd(&num_g[z * (DH * DE) + tid], num_part[tid]);
    if (tid < DH) atomicAdd(&den_g[z * DH + tid], den_part[tid]);
}

// out[z] = normed( num[z]/den[z,h] , over all 1024 )
__global__ __launch_bounds__(1024)
void final_kernel(const float* __restrict__ num_g,
                  const float* __restrict__ den_g,
                  float* __restrict__ out) {
    __shared__ float red[32];
    const int z = blockIdx.x, i = threadIdx.x;
    float t = num_g[z * 1024 + i] / fmaxf(den_g[z * DH + (i >> 6)], 1e-38f);
    float s = t;
    #pragma unroll
    for (int off = 16; off >= 1; off >>= 1) s += __shfl_xor(s, off, 32);
    if ((i & 31) == 0) red[i >> 5] = s;
    __syncthreads();
    if (i < 32) {
        float v = red[i];
        #pragma unroll
        for (int off = 16; off >= 1; off >>= 1) v += __shfl_xor(v, off, 32);
        if (i == 0) red[0] = v;
    }
    __syncthreads();
    out[z * 1024 + i] = t / fmaxf(red[0], 1e-38f);
}

extern "C" void kernel_launch(void* const* d_in, const int* in_sizes, int n_in,
                              void* d_out, int out_size, void* d_ws, size_t ws_size,
                              hipStream_t stream) {
    const float*         x    = (const float*)d_in[0];
    const unsigned char* mask = (const unsigned char*)d_in[1];
    const float*         Wq   = (const float*)d_in[2];
    const float*         bq   = (const float*)d_in[3];
    const float*         Wkv  = (const float*)d_in[4];
    const float*         bkv  = (const float*)d_in[5];
    float*               out  = (float*)d_out;

    // workspace: [num 32768][den 512][q 32768][WkvT 2M bf16]
    float* num_g = (float*)d_ws;
    float* den_g = num_g + 32768;
    float* q_g   = den_g + 512;
    bf16*  wkvT  = (bf16*)(q_g + 32768);

    zero_kernel<<<(33280 + 255) / 256, 256, 0, stream>>>(num_g, 33280);
    dim3 tgrid(64, 32);
    cvtT_kernel<<<tgrid, 256, 0, stream>>>(Wkv, wkvT);
    dim3 qgrid(DZ, 4);
    q_kernel<<<qgrid, 256, 0, stream>>>(x, Wq, bq, q_g);
    kv_attn_kernel<<<(DZ * DK) / TM, 1024, SMEM_TOTAL, stream>>>(
        x, mask, wkvT, bkv, q_g, num_g, den_g);
    final_kernel<<<DZ, 1024, 0, stream>>>(num_g, den_g, out);
}
